// CustomGNNForecaster_30820685316435
// MI455X (gfx1250) — matrix operand order repaired and checked by reading, not compiled, not run on hardware
//
#include <hip/hip_runtime.h>
#include <math.h>

typedef __attribute__((ext_vector_type(16))) _Float16 v16h;
typedef __attribute__((ext_vector_type(8)))  float    v8f;

#define CDIV(a, b) (((a) + (b) - 1) / (b))

// ---------------------------------------------------------------------------
// WMMA fragment loaders (layouts per CDNA5 ISA 7.12.2, wave32)
// ---------------------------------------------------------------------------

// A: 16x32 f16 tile, row-major source [M x lda].
// lane L (half = L>>4): row = row0 + (L&15); elems 0..7 = K kb..kb+7,
// elems 8..15 = K kb+16..kb+23, kb = k0 + half*8.  Two 16B contiguous loads.
__device__ __forceinline__ v16h load_a_frag(const _Float16* A, int lda, int M,
                                            int row0, int k0, int lane) {
    int r = row0 + (lane & 15);
    if (r >= M) r = M - 1;                 // clamp (safe, stores are guarded)
    int kb = k0 + ((lane >> 4) << 3);
    const _Float16* p = A + (size_t)r * lda + kb;
    v16h a;
    uint4* pa = (uint4*)&a;
    pa[0] = *(const uint4*)(p);
    pa[1] = *(const uint4*)(p + 16);
    return a;
}

// B fragments pre-packed per tile: Bp[(((kt*ntiles)+nt)*32 + lane)*16 + i].
// Each lane: two 16B reads, at p and at p+16 halves (32B apart).
__device__ __forceinline__ v16h load_b_frag_packed(const _Float16* Bp, int ntiles,
                                                   int kt, int nt, int lane) {
    const _Float16* p = Bp + ((size_t)(((kt * ntiles) + nt) * 32 + lane) << 4);
    v16h b;
    uint4* pb = (uint4*)&b;
    pb[0] = *(const uint4*)(p);
    pb[1] = *(const uint4*)(p + 16);
    return b;
}

// ---------------------------------------------------------------------------
// Generic f16 GEMM: C[M x Ncols](f32) = A[M x lda](f16) * B(frag-packed f16).
// KT (1 or 2) and NTG are compile-time so A fragments live in fixed VGPRs
// (no v_movrels) and all inner loops fully unroll.  Each wave owns one
// 16-row m-tile and NTG n-tiles, reusing its A fragments across the group.
// ---------------------------------------------------------------------------
template <int KT, int NTG>
__global__ __launch_bounds__(256) void gemm_f16(const _Float16* __restrict__ A,
                                                const _Float16* __restrict__ Bp,
                                                float* __restrict__ C,
                                                int M, int Ncols, int lda) {
    int wid    = (blockIdx.x * blockDim.x + threadIdx.x) >> 5;
    int lane   = threadIdx.x & 31;
    int ntiles = Ncols >> 4;
    int groups = ntiles / NTG;
    int mtiles = (M + 15) >> 4;
    int mt = wid / groups;
    int g  = wid - mt * groups;
    if (mt >= mtiles) return;              // wave-uniform exit: EXEC stays full

    v16h a0 = load_a_frag(A, lda, M, mt * 16, 0, lane);
    v16h a1;
    if (KT == 2) a1 = load_a_frag(A, lda, M, mt * 16, 32, lane);

    bool full  = (mt * 16 + 16) <= M;      // wave-uniform
    int  colr  = lane & 15;
    int  rbase = mt * 16 + ((lane >> 4) << 3);

#pragma unroll
    for (int j = 0; j < NTG; ++j) {
        int nt = g * NTG + j;
        v8f acc = {};
        {
            v16h b = load_b_frag_packed(Bp, ntiles, 0, nt, lane);
            acc = __builtin_amdgcn_wmma_f32_16x16x32_f16(false, a0, false, b,
                                                         (short)0, acc, false, false);
        }
        if (KT == 2) {
            v16h b = load_b_frag_packed(Bp, ntiles, 1, nt, lane);
            acc = __builtin_amdgcn_wmma_f32_16x16x32_f16(false, a1, false, b,
                                                         (short)0, acc, false, false);
        }
        int col = nt * 16 + colr;
        float* Cp = C + (size_t)rbase * Ncols + col;
        if (full) {
#pragma unroll
            for (int r = 0; r < 8; ++r) Cp[(size_t)r * Ncols] = acc[r];
        } else {
#pragma unroll
            for (int r = 0; r < 8; ++r)
                if (rbase + r < M) Cp[(size_t)r * Ncols] = acc[r];
        }
    }
}

// ---------------------------------------------------------------------------
// Setup kernels
// ---------------------------------------------------------------------------
__global__ void fill_ones(float* p, int n) {
    int i = blockIdx.x * blockDim.x + threadIdx.x;
    if (i < n) p[i] = 1.0f;
}
__global__ void deg_accum(const int* __restrict__ dst, float* deg, int E) {
    int e = blockIdx.x * blockDim.x + threadIdx.x;
    if (e < E) atomicAdd(&deg[dst[e]], 1.0f);
}
__global__ void rsqrt_inplace(float* p, int n) {
    int i = blockIdx.x * blockDim.x + threadIdx.x;
    if (i < n) p[i] = rsqrtf(p[i]);
}

// Repack weight into WMMA B-fragment order.
// transpose==0: W is (K, Nout) row-major (h @ W).
// transpose==1: W is (Nout, K) row-major, used as x @ W.T (torch GRU layout).
// Output layout matches load_b_frag_packed: element i of lane `lane` of tile
// (kt, nt) corresponds to B[k][c], k = kt*32 + (lane>>4)*8 + (i&7) + (i>>3)*16, c = nt*16+(lane&15).
__global__ void pack_frag_b(const float* __restrict__ W, _Float16* __restrict__ Bp,
                            int K, int Nout, int transpose) {
    int idx = blockIdx.x * blockDim.x + threadIdx.x;
    if (idx >= K * Nout) return;
    int i    = idx & 15;
    int lane = (idx >> 4) & 31;
    int tile = idx >> 9;
    int ntiles = Nout >> 4;
    int kt = tile / ntiles;
    int nt = tile - kt * ntiles;
    int k = kt * 32 + ((lane >> 4) << 3) + (i & 7) + ((i >> 3) << 4);
    int c = nt * 16 + (lane & 15);
    float v = transpose ? W[(size_t)c * K + k] : W[(size_t)k * Nout + c];
    Bp[idx] = (_Float16)v;
}

// ---------------------------------------------------------------------------
// GCN layer 1 dense part: hw1[n, j<32] = sum_{k<5} x[n,t,k] * W_g1[k,j]
// ---------------------------------------------------------------------------
__global__ void gcn1_mm(const float* __restrict__ x, const float* __restrict__ Wg1,
                        float* __restrict__ hw1, int Nn, int t) {
    int idx = blockIdx.x * blockDim.x + threadIdx.x;
    if (idx >= Nn * 32) return;
    int n = idx >> 5, j = idx & 31;
    const float* xp = x + (size_t)n * 60 + t * 5;
    float acc = 0.f;
#pragma unroll
    for (int k = 0; k < 5; ++k) acc += xp[k] * Wg1[k * 32 + j];
    hw1[idx] = acc;
}

// scatter: agg[dst, f] += hw[src, f] * dinv[src] * dinv[dst];  F = 1<<fsh
__global__ void scatter_agg(const int* __restrict__ src, const int* __restrict__ dst,
                            const float* __restrict__ dinv,
                            const float* __restrict__ hw, float* __restrict__ agg,
                            int E, int fsh) {
    long long tid = (long long)blockIdx.x * blockDim.x + threadIdx.x;
    if (tid >= ((long long)E << fsh)) return;
    int e = (int)(tid >> fsh);
    int f = (int)(tid & ((1 << fsh) - 1));
    int s = src[e], d = dst[e];
    float en = dinv[s] * dinv[d];
    atomicAdd(&agg[((size_t)d << fsh) + f], hw[((size_t)s << fsh) + f] * en);
}

// out = relu(agg + hw * dinv^2 + bias) cast to f16
__global__ void gcn_post(const float* __restrict__ hw, const float* __restrict__ agg,
                         const float* __restrict__ dinv, const float* __restrict__ bias,
                         _Float16* __restrict__ out, int Nn, int fsh) {
    int idx = blockIdx.x * blockDim.x + threadIdx.x;
    if (idx >= (Nn << fsh)) return;
    int n = idx >> fsh, f = idx & ((1 << fsh) - 1);
    float di = dinv[n];
    float v = agg[idx] + hw[idx] * di * di + bias[f];
    out[idx] = (_Float16)fmaxf(v, 0.f);
}

// ---------------------------------------------------------------------------
// GRU cell: torch semantics, gi/gh are [N,192] (r|z|c)
// ---------------------------------------------------------------------------
__global__ void gru_cell(const float* __restrict__ gi, const float* __restrict__ gh,
                         const float* __restrict__ bih, const float* __restrict__ bhh,
                         float* __restrict__ h, _Float16* __restrict__ h16,
                         _Float16* __restrict__ rnn_slot, int Nn) {
    int idx = blockIdx.x * blockDim.x + threadIdx.x;
    if (idx >= Nn * 64) return;
    int n = idx >> 6, d = idx & 63;
    size_t base = (size_t)n * 192;
    float ir = gi[base + d]        + bih[d];
    float iz = gi[base + 64 + d]   + bih[64 + d];
    float ic = gi[base + 128 + d]  + bih[128 + d];
    float hr = gh[base + d]        + bhh[d];
    float hz = gh[base + 64 + d]   + bhh[64 + d];
    float hc = gh[base + 128 + d]  + bhh[128 + d];
    float r = 1.f / (1.f + expf(-(ir + hr)));
    float z = 1.f / (1.f + expf(-(iz + hz)));
    float c = tanhf(ic + r * hc);
    float hp = h[idx];
    float hn = (1.f - z) * c + z * hp;
    h[idx]   = hn;
    h16[idx] = (_Float16)hn;
    if (rnn_slot) rnn_slot[idx] = (_Float16)hn;
}

// ---------------------------------------------------------------------------
// Attention scores: per 16-row tile, score[row] = sum_c tanh((A@Wa1)[row,c]+b[c])*w2[c]
// one wave handles 16 rows x all 64 cols (4 WMMA column tiles, A frags reused)
// ---------------------------------------------------------------------------
__global__ __launch_bounds__(256) void attn_scores(const _Float16* __restrict__ A,
                                                   const _Float16* __restrict__ Bp,
                                                   const float* __restrict__ ba1,
                                                   const float* __restrict__ wa2,
                                                   float* __restrict__ scores, int R) {
    int wid  = (blockIdx.x * blockDim.x + threadIdx.x) >> 5;
    int lane = threadIdx.x & 31;
    int rtiles = (R + 15) >> 4;
    if (wid >= rtiles) return;

    v16h a0 = load_a_frag(A, 64, R, wid * 16, 0, lane);
    v16h a1 = load_a_frag(A, 64, R, wid * 16, 32, lane);
    float part[8];
#pragma unroll
    for (int r = 0; r < 8; ++r) part[r] = 0.f;

#pragma unroll
    for (int g = 0; g < 4; ++g) {
        v8f acc = {};
        v16h b0 = load_b_frag_packed(Bp, 4, 0, g, lane);
        acc = __builtin_amdgcn_wmma_f32_16x16x32_f16(false, a0, false, b0,
                                                     (short)0, acc, false, false);
        v16h b1 = load_b_frag_packed(Bp, 4, 1, g, lane);
        acc = __builtin_amdgcn_wmma_f32_16x16x32_f16(false, a1, false, b1,
                                                     (short)0, acc, false, false);
        int col = g * 16 + (lane & 15);
        float bb = ba1[col], w2 = wa2[col];
#pragma unroll
        for (int r = 0; r < 8; ++r) part[r] += tanhf(acc[r] + bb) * w2;
    }
    // reduce across 16 lanes holding the same rows (xor 1,2,4,8 stays in half)
#pragma unroll
    for (int r = 0; r < 8; ++r) {
        float v = part[r];
        for (int s = 1; s < 16; s <<= 1) v += __shfl_xor(v, s, 32);
        part[r] = v;
    }
    if ((lane & 15) == 0) {
        int rbase = wid * 16 + ((lane >> 4) << 3);
#pragma unroll
        for (int r = 0; r < 8; ++r) {
            int row = rbase + r;
            if (row < R) scores[row] = part[r];
        }
    }
}

// ---------------------------------------------------------------------------
// softmax over T, weighted sum, output MLP (one block of 64 threads per node)
// ---------------------------------------------------------------------------
__global__ void attn_out(const float* __restrict__ scores,
                         const _Float16* __restrict__ rnn,  // [T, N, 64]
                         const float* __restrict__ Wo1, const float* __restrict__ bo1,
                         const float* __restrict__ Wo2, const float* __restrict__ bo2,
                         float* __restrict__ out, int Nn) {
    int n = blockIdx.x;
    int d = threadIdx.x;    // 0..63
    __shared__ float att[64];
    __shared__ float hid[32];

    float s[12];
    float mx = -1e30f;
#pragma unroll
    for (int t = 0; t < 12; ++t) {
        s[t] = scores[(size_t)t * Nn + n];
        mx = fmaxf(mx, s[t]);
    }
    float sum = 0.f;
#pragma unroll
    for (int t = 0; t < 12; ++t) { s[t] = expf(s[t] - mx); sum += s[t]; }
    float inv = 1.f / sum;
    float a = 0.f;
#pragma unroll
    for (int t = 0; t < 12; ++t)
        a += s[t] * (float)rnn[((size_t)t * Nn + n) * 64 + d];
    att[d] = a * inv;
    __syncthreads();
    if (d < 32) {
        float acc = bo1[d];
#pragma unroll
        for (int k = 0; k < 64; ++k) acc += att[k] * Wo1[k * 32 + d];
        hid[d] = fmaxf(acc, 0.f);
    }
    __syncthreads();
    if (d == 0) {
        float acc = bo2[0];
#pragma unroll
        for (int k = 0; k < 32; ++k) acc += hid[k] * Wo2[k];
        out[n] = acc;
    }
}

// ---------------------------------------------------------------------------
// Host side
// ---------------------------------------------------------------------------
static inline char* carve(char*& p, size_t bytes) {
    char* r = p;
    p += (bytes + 255) & ~(size_t)255;
    return r;
}

static void launch_gemm(const _Float16* A, int lda, const _Float16* Bp, float* C,
                        int M, int Ncols, int Kt, hipStream_t s) {
    const int NTG = 4;
    int mt = (M + 15) / 16;
    int groups = (Ncols / 16) / NTG;
    long long waves = (long long)mt * groups;
    int blocks = (int)((waves * 32 + 255) / 256);
    if (Kt == 1)
        gemm_f16<1, NTG><<<blocks, 256, 0, s>>>(A, Bp, C, M, Ncols, lda);
    else
        gemm_f16<2, NTG><<<blocks, 256, 0, s>>>(A, Bp, C, M, Ncols, lda);
}

extern "C" void kernel_launch(void* const* d_in, const int* in_sizes, int n_in,
                              void* d_out, int out_size, void* d_ws, size_t ws_size,
                              hipStream_t stream) {
    const float* x     = (const float*)d_in[0];
    const int*   ei    = (const int*)d_in[1];
    const float* W_g1  = (const float*)d_in[2];
    const float* b_g1  = (const float*)d_in[3];
    const float* W_g2  = (const float*)d_in[4];
    const float* b_g2  = (const float*)d_in[5];
    const float* W_ih0 = (const float*)d_in[6];
    const float* W_hh0 = (const float*)d_in[7];
    const float* b_ih0 = (const float*)d_in[8];
    const float* b_hh0 = (const float*)d_in[9];
    const float* W_ih1 = (const float*)d_in[10];
    const float* W_hh1 = (const float*)d_in[11];
    const float* b_ih1 = (const float*)d_in[12];
    const float* b_hh1 = (const float*)d_in[13];
    const float* W_a1  = (const float*)d_in[14];
    const float* b_a1  = (const float*)d_in[15];
    const float* w_a2  = (const float*)d_in[16];
    const float* W_o1  = (const float*)d_in[17];
    const float* b_o1  = (const float*)d_in[18];
    const float* W_o2  = (const float*)d_in[19];
    const float* b_o2  = (const float*)d_in[20];

    const int Nn = in_sizes[0] / 60;   // (N, 12, 5)
    const int Ee = in_sizes[1] / 2;    // (2, E)
    const int* src = ei;
    const int* dst = ei + Ee;

    char* p = (char*)d_ws;
    float*     dinv   = (float*)carve(p, (size_t)Nn * 4);
    float*     hw1    = (float*)carve(p, (size_t)Nn * 32 * 4);
    float*     agg1   = (float*)carve(p, (size_t)Nn * 32 * 4);
    _Float16*  h1_16  = (_Float16*)carve(p, (size_t)Nn * 32 * 2);
    float*     hw2    = (float*)carve(p, (size_t)Nn * 64 * 4);
    float*     agg2   = (float*)carve(p, (size_t)Nn * 64 * 4);
    _Float16*  x0_16  = (_Float16*)carve(p, (size_t)Nn * 64 * 2);
    float*     hl0    = (float*)carve(p, (size_t)Nn * 64 * 4);
    _Float16*  hl0_16 = (_Float16*)carve(p, (size_t)Nn * 64 * 2);
    float*     hl1    = (float*)carve(p, (size_t)Nn * 64 * 4);
    _Float16*  hl1_16 = (_Float16*)carve(p, (size_t)Nn * 64 * 2);
    float*     gi     = (float*)carve(p, (size_t)Nn * 192 * 4);
    float*     gh     = (float*)carve(p, (size_t)Nn * 192 * 4);
    _Float16*  rnn16  = (_Float16*)carve(p, (size_t)12 * Nn * 64 * 2);
    float*     scores = (float*)carve(p, (size_t)12 * Nn * 4);
    _Float16*  Bg2    = (_Float16*)carve(p, 32 * 64 * 2);
    _Float16*  Bih0   = (_Float16*)carve(p, 64 * 192 * 2);
    _Float16*  Bhh0   = (_Float16*)carve(p, 64 * 192 * 2);
    _Float16*  Bih1   = (_Float16*)carve(p, 64 * 192 * 2);
    _Float16*  Bhh1   = (_Float16*)carve(p, 64 * 192 * 2);
    _Float16*  Ba1    = (_Float16*)carve(p, 64 * 64 * 2);

    // degree / normalization (deg starts at 1 for self-loop, then rsqrt in place)
    fill_ones<<<CDIV(Nn, 256), 256, 0, stream>>>(dinv, Nn);
    deg_accum<<<CDIV(Ee, 256), 256, 0, stream>>>(dst, dinv, Ee);
    rsqrt_inplace<<<CDIV(Nn, 256), 256, 0, stream>>>(dinv, Nn);

    // weight repack into WMMA fragment order (f16)
    pack_frag_b<<<CDIV(32 * 64, 256), 256, 0, stream>>>(W_g2, Bg2, 32, 64, 0);
    pack_frag_b<<<CDIV(64 * 192, 256), 256, 0, stream>>>(W_ih0, Bih0, 64, 192, 1);
    pack_frag_b<<<CDIV(64 * 192, 256), 256, 0, stream>>>(W_hh0, Bhh0, 64, 192, 1);
    pack_frag_b<<<CDIV(64 * 192, 256), 256, 0, stream>>>(W_ih1, Bih1, 64, 192, 1);
    pack_frag_b<<<CDIV(64 * 192, 256), 256, 0, stream>>>(W_hh1, Bhh1, 64, 192, 1);
    pack_frag_b<<<CDIV(64 * 64, 256), 256, 0, stream>>>(W_a1, Ba1, 64, 64, 0);

    // GRU hidden state init
    hipMemsetAsync(hl0, 0, (size_t)Nn * 64 * 4, stream);
    hipMemsetAsync(hl1, 0, (size_t)Nn * 64 * 4, stream);
    hipMemsetAsync(hl0_16, 0, (size_t)Nn * 64 * 2, stream);
    hipMemsetAsync(hl1_16, 0, (size_t)Nn * 64 * 2, stream);

    for (int t = 0; t < 12; ++t) {
        // GCN layer 1 (K=5, scalar)
        gcn1_mm<<<CDIV(Nn * 32, 256), 256, 0, stream>>>(x, W_g1, hw1, Nn, t);
        hipMemsetAsync(agg1, 0, (size_t)Nn * 32 * 4, stream);
        scatter_agg<<<(int)CDIV((long long)Ee * 32, 256), 256, 0, stream>>>(
            src, dst, dinv, hw1, agg1, Ee, 5);
        gcn_post<<<CDIV(Nn * 32, 256), 256, 0, stream>>>(hw1, agg1, dinv, b_g1,
                                                         h1_16, Nn, 5);
        // GCN layer 2 (WMMA GEMM, K=32)
        launch_gemm(h1_16, 32, Bg2, hw2, Nn, 64, 1, stream);
        hipMemsetAsync(agg2, 0, (size_t)Nn * 64 * 4, stream);
        scatter_agg<<<(int)CDIV((long long)Ee * 64, 256), 256, 0, stream>>>(
            src, dst, dinv, hw2, agg2, Ee, 6);
        gcn_post<<<CDIV(Nn * 64, 256), 256, 0, stream>>>(hw2, agg2, dinv, b_g2,
                                                         x0_16, Nn, 6);
        // GRU layer 0
        launch_gemm(x0_16, 64, Bih0, gi, Nn, 192, 2, stream);
        launch_gemm(hl0_16, 64, Bhh0, gh, Nn, 192, 2, stream);
        gru_cell<<<CDIV(Nn * 64, 256), 256, 0, stream>>>(gi, gh, b_ih0, b_hh0,
                                                         hl0, hl0_16,
                                                         (_Float16*)nullptr, Nn);
        // GRU layer 1
        launch_gemm(hl0_16, 64, Bih1, gi, Nn, 192, 2, stream);
        launch_gemm(hl1_16, 64, Bhh1, gh, Nn, 192, 2, stream);
        gru_cell<<<CDIV(Nn * 64, 256), 256, 0, stream>>>(gi, gh, b_ih1, b_hh1,
                                                         hl1, hl1_16,
                                                         rnn16 + (size_t)t * Nn * 64,
                                                         Nn);
    }

    // temporal attention scores (WMMA, fused tanh + dot + reduce)
    int R = 12 * Nn;
    int rtiles = CDIV(R, 16);
    attn_scores<<<(int)CDIV((long long)rtiles * 32, 256), 256, 0, stream>>>(
        rnn16, Ba1, b_a1, w_a2, scores, R);

    // softmax + weighted sum + output MLP
    attn_out<<<Nn, 64, 0, stream>>>(scores, rnn16, W_o1, b_o1, W_o2, b_o2,
                                    (float*)d_out, Nn);
}
